// SAMBA_65730179498082
// MI455X (gfx1250) — matrix operand order repaired and checked by reading, not compiled, not run on hardware
//
#include <hip/hip_runtime.h>
#include <hip/hip_bf16.h>
#include <math.h>

// ---------------------------------------------------------------------------
// SAMBA forward for MI455X (gfx1250, wave32).
// Dense projections: f32 operands are converted to f16 once (A: streaming
// pass; B/weights: packed into exact WMMA fragment layout), then a branch-free
// GEMM runs 4x v_wmma_f32_16x16x32_f16 per K-chunk per wave (64x16 C strip)
// with only b128 loads in the inner loop. SSM scans use one wave per 32-wide
// state row with shfl reductions. Workspace: static bump layout, L2-resident.
// ---------------------------------------------------------------------------

typedef __attribute__((ext_vector_type(16))) _Float16 v16h;
typedef __attribute__((ext_vector_type(8)))  _Float16 v8h;
typedef __attribute__((ext_vector_type(8)))  float    v8f;

union V16U { v16h v; struct { v8h lo, hi; } p; };

__device__ __forceinline__ float siluf(float x) { return x / (1.f + __expf(-x)); }

// ---- f32 -> f16 streaming convert ------------------------------------------
__global__ void cvt_f16_kernel(const float* __restrict__ in, _Float16* __restrict__ out, long n)
{
    long i = (long)blockIdx.x * blockDim.x + threadIdx.x;
    if (i < n) out[i] = (_Float16)in[i];
}

// ---- pack B (f32 row-major KxN) into per-chunk WMMA fragment layout --------
// Layout: [batch][tileN][kchunk][lane(32)][16 halves]; lane holds col=lane%16,
// K = kc*32 + (lane/16)*8 + 2j (+1) + (j>=4 ? 8 : 0). OOB K/N positions are zero-filled.
__global__ void pack_b_kernel(const float* __restrict__ Bw, _Float16* __restrict__ out,
                              int N, int K, int kchunks, long sB, long sOut)
{
    int lane = threadIdx.x & 31;
    int tn = blockIdx.x, kc = blockIdx.y, bz = blockIdx.z;
    const float* Bb = Bw + (long)bz * sB;
    _Float16* op = out + (long)bz * sOut + ((long)tn * kchunks + kc) * 512 + lane * 16;
    int half = lane >> 4, l16 = lane & 15;
    int col = tn * 16 + l16;
#pragma unroll
    for (int j = 0; j < 8; ++j) {
        int kb = kc * 32 + half * 8 + 2 * j + (j >= 4 ? 8 : 0);
        float b0 = 0.f, b1 = 0.f;
        if (col < N) {
            if (kb     < K) b0 = Bb[(long)kb * N + col];
            if (kb + 1 < K) b1 = Bb[(long)(kb + 1) * N + col];
        }
        op[2 * j]     = (_Float16)b0;
        op[2 * j + 1] = (_Float16)b1;
    }
}

// ---------------- WMMA GEMM: C = act(Ah[M,K] @ Bpack + bias) ----------------
// One wave per 64x16 C strip; grid (tilesN, M/64, batch). M must be a multiple
// of 64 (true for every call site). A fragment = two v8h loads per subtile at
// halves [k0 + half*8, +8) and [k0+16+half*8, +8); B fragment = one contiguous
// 32-byte read per lane from the packed buffer.
__global__ void wmma_gemm_f16_kernel(const _Float16* __restrict__ Ah,
                                     const _Float16* __restrict__ Bp,
                                     const float* __restrict__ bias, float* __restrict__ C,
                                     int M, int N, int K, int kchunks,
                                     long sA, long sB, long sC, int act)
{
    int lane = threadIdx.x & 31;
    int half = lane >> 4;
    int l16  = lane & 15;
    int col     = blockIdx.x * 16 + l16;
    int rowBase = blockIdx.y * 64;
    const _Float16* a0p = Ah + (long)blockIdx.z * sA + (long)(rowBase + l16) * K + half * 8;
    const _Float16* a1p = a0p + (long)16 * K;
    const _Float16* a2p = a0p + (long)32 * K;
    const _Float16* a3p = a0p + (long)48 * K;
    const _Float16* bp  = Bp + (long)blockIdx.z * sB + ((long)blockIdx.x * kchunks) * 512 + lane * 16;
    float* Cb = C + (long)blockIdx.z * sC;

    v8f acc0 = {}, acc1 = {}, acc2 = {}, acc3 = {};
    for (int kc = 0; kc < kchunks; ++kc) {
        int k0 = kc * 32;
        V16U bf, a0, a1, a2, a3;
        bf.p.lo = *(const v8h*)(bp);
        bf.p.hi = *(const v8h*)(bp + 8);
        a0.p.lo = *(const v8h*)(a0p + k0);  a0.p.hi = *(const v8h*)(a0p + k0 + 16);
        a1.p.lo = *(const v8h*)(a1p + k0);  a1.p.hi = *(const v8h*)(a1p + k0 + 16);
        a2.p.lo = *(const v8h*)(a2p + k0);  a2.p.hi = *(const v8h*)(a2p + k0 + 16);
        a3.p.lo = *(const v8h*)(a3p + k0);  a3.p.hi = *(const v8h*)(a3p + k0 + 16);
        acc0 = __builtin_amdgcn_wmma_f32_16x16x32_f16(false, a0.v, false, bf.v, (short)0, acc0, false, false);
        acc1 = __builtin_amdgcn_wmma_f32_16x16x32_f16(false, a1.v, false, bf.v, (short)0, acc1, false, false);
        acc2 = __builtin_amdgcn_wmma_f32_16x16x32_f16(false, a2.v, false, bf.v, (short)0, acc2, false, false);
        acc3 = __builtin_amdgcn_wmma_f32_16x16x32_f16(false, a3.v, false, bf.v, (short)0, acc3, false, false);
        bp += 512;
    }
    if (col < N) {
        float bv = bias ? bias[col] : 0.f;
#pragma unroll
        for (int v = 0; v < 8; ++v) {            // D layout: row = vgpr + 8*(lane/16)
            int r0 = rowBase + v + 8 * half;
            float x0 = acc0[v] + bv, x1 = acc1[v] + bv, x2 = acc2[v] + bv, x3 = acc3[v] + bv;
            if (act == 1) { x0 = siluf(x0); x1 = siluf(x1); x2 = siluf(x2); x3 = siluf(x3); }
            Cb[(long)r0 * N + col]        = x0;
            Cb[(long)(r0 + 16) * N + col] = x1;
            Cb[(long)(r0 + 32) * N + col] = x2;
            Cb[(long)(r0 + 48) * N + col] = x3;
        }
    }
}

// ---------------- direct NCH conv (k=1/3/7), writes into channel window -----
__global__ void conv1d_kernel(const float* __restrict__ in, const float* __restrict__ w,
                              const float* __restrict__ bias, float* __restrict__ out,
                              int Ci, int L, int Kw, int padL, int CoTotal, int coOff)
{
    int t  = blockIdx.x * blockDim.x + threadIdx.x;
    int co = blockIdx.y;
    int b  = blockIdx.z;
    if (t >= L) return;
    float s = bias[co];
    for (int ci = 0; ci < Ci; ++ci) {
        const float* ip = in + ((long)b * Ci + ci) * L;
        const float* wp = w + ((long)co * Ci + ci) * Kw;
        for (int k = 0; k < Kw; ++k) {
            int tt = t + k - padL;
            if (tt >= 0 && tt < L) s += wp[k] * ip[tt];
        }
    }
    out[((long)b * CoTotal + (co + coOff)) * L + t] = s;
}

// ---------------- depthwise causal conv (k=4, pad (3,0)) + SiLU -------------
__global__ void dwconv_silu_kernel(const float* __restrict__ in, long ldin,
                                   float* __restrict__ out, int T, int Cc,
                                   const float* __restrict__ w, const float* __restrict__ wb,
                                   long wStride, long bStride, int batchPerHead, long total)
{
    long idx = (long)blockIdx.x * blockDim.x + threadIdx.x;
    if (idx >= total) return;
    int c  = (int)(idx % Cc);
    long r = idx / Cc;
    int t  = (int)(r % T);
    int nb = (int)(r / T);
    int head = nb / batchPerHead;
    const float* wp   = w + head * wStride + (long)c * 4;
    const float* base = in + (long)nb * T * ldin;
    float s = wb[head * bStride + c];
#pragma unroll
    for (int k = 0; k < 4; ++k) {
        int tt = t - 3 + k;
        if (tt >= 0) s += wp[k] * base[(long)tt * ldin + c];
    }
    out[idx] = siluf(s);
}

// ---------------- dt = softplus(raw + bias), dA = exp(-exp(A_log)*dt) -------
__global__ void dt_prep_kernel(const float* __restrict__ zxb, int dinp, int dtoff,
                               const float* __restrict__ dt_bias, const float* __restrict__ A_log,
                               float* __restrict__ dts, float* __restrict__ dAa,
                               int T, int nh, long pStride, int batchPerHead, long total)
{
    long idx = (long)blockIdx.x * blockDim.x + threadIdx.x;
    if (idx >= total) return;
    int h  = (int)(idx % nh);
    long r = idx / nh;
    int nb = (int)(r / T);
    int head = nb / batchPerHead;
    float raw = zxb[r * dinp + dtoff + h] + dt_bias[head * pStride + h];
    float sp  = raw > 20.f ? raw : log1pf(__expf(raw));
    dts[idx] = sp;
    dAa[idx] = __expf(-__expf(A_log[head * pStride + h]) * sp);
}

// ---------------- sequential SSM scan ---------------------------------------
__global__ void ssm_scan_kernel(const float* __restrict__ dAa, const float* __restrict__ dts,
                                const float* __restrict__ xbc, int ldx, int dinner,
                                const float* __restrict__ Dp, long DStride,
                                float* __restrict__ y, int T, int nh, int hd, int batchPerHead)
{
    const int NS = 32;
    int bh = blockIdx.x;
    int h  = bh % nh;
    int nb = bh / nh;
    int head = nb / batchPerHead;
    int n  = threadIdx.x;
    int py = threadIdx.y;
    int RP = hd >> 3;
    float hstate[8];
#pragma unroll
    for (int i = 0; i < 8; ++i) hstate[i] = 0.f;
    float Dv = Dp[head * DStride + h];
    const float* rowbase = xbc + (long)nb * T * ldx;
    for (int t = 0; t < T; ++t) {
        const float* rrow = rowbase + (long)t * ldx;
        long sidx = ((long)nb * T + t) * nh + h;
        float a  = dAa[sidx];
        float d  = dts[sidx];
        float Bn = rrow[dinner + n];
        float Cn = rrow[dinner + NS + n];
        float dB = d * Bn;
        for (int i = 0; i < RP; ++i) {
            int p = py + i * 8;
            float xv = rrow[h * hd + p];
            float hs = a * hstate[i] + dB * xv;
            hstate[i] = hs;
            float part = hs * Cn;
            part += __shfl_xor(part, 16, 32);
            part += __shfl_xor(part, 8, 32);
            part += __shfl_xor(part, 4, 32);
            part += __shfl_xor(part, 2, 32);
            part += __shfl_xor(part, 1, 32);
            if (n == 0)
                y[((long)nb * T + t) * (long)(nh * hd) + h * hd + p] = part + Dv * xv;
        }
    }
}

// ---------------- y = rmsnorm(y * silu(z)) * norm_w (in-place) --------------
__global__ void gate_rmsnorm_kernel(float* __restrict__ y, const float* __restrict__ zxb,
                                    int dinp, const float* __restrict__ norm_w, long nwStride,
                                    int T, int dinner, int batchPerHead)
{
    long rowid = blockIdx.x;
    int nb = (int)(rowid / T);
    int head = nb / batchPerHead;
    int c = threadIdx.x;
    float z = zxb[rowid * (long)dinp + c];
    float v = y[rowid * (long)dinner + c] * siluf(z);
    __shared__ float red[256];
    red[c] = v * v;
    __syncthreads();
    for (int s = blockDim.x >> 1; s > 0; s >>= 1) {
        if (c < s) red[c] += red[c + s];
        __syncthreads();
    }
    float inv = rsqrtf(red[0] / dinner + 1e-5f);
    y[rowid * (long)dinner + c] = v * inv * norm_w[head * nwStride + c];
}

// ---------------- elementwise / layout helpers ------------------------------
__global__ void add_kernel(const float* a, const float* b, float* o, long n)
{
    long i = (long)blockIdx.x * blockDim.x + threadIdx.x;
    if (i < n) o[i] = a[i] + b[i];
}

__global__ void bct_to_btc_kernel(const float* in, float* out, int C, int T, long total)
{
    long idx = (long)blockIdx.x * blockDim.x + threadIdx.x;
    if (idx >= total) return;
    int c  = (int)(idx % C);
    long r = idx / C;
    int t  = (int)(r % T);
    long b = r / T;
    out[idx] = in[(b * C + c) * (long)T + t];
}

__global__ void btc_to_bct_kernel(const float* in, float* out, int C, int T, long total)
{
    long idx = (long)blockIdx.x * blockDim.x + threadIdx.x;
    if (idx >= total) return;
    int t  = (int)(idx % T);
    long r = idx / T;
    int c  = (int)(r % C);
    long b = r / C;
    out[idx] = in[(b * T + t) * (long)C + c];
}

__global__ void maxpool2_kernel(const float* in, float* out, int Lout, long total)
{
    long idx = (long)blockIdx.x * blockDim.x + threadIdx.x;
    if (idx >= total) return;
    int l   = (int)(idx % Lout);
    long bc = idx / Lout;
    const float* p = in + bc * (long)(2 * Lout) + 2 * l;
    out[idx] = fmaxf(p[0], p[1]);
}

__global__ void interp_kernel(const float* in, float* out, int Lin, int Lout, long total)
{
    long idx = (long)blockIdx.x * blockDim.x + threadIdx.x;
    if (idx >= total) return;
    int l   = (int)(idx % Lout);
    long bc = idx / Lout;
    float src = (l + 0.5f) * ((float)Lin / (float)Lout) - 0.5f;
    src = fminf(fmaxf(src, 0.f), (float)(Lin - 1));
    int i0 = (int)floorf(src);
    int i1 = min(i0 + 1, Lin - 1);
    float w = src - (float)i0;
    out[idx] = in[bc * Lin + i0] * (1.f - w) + in[bc * Lin + i1] * w;
}

__global__ void copy_chan_kernel(const float* in, float* out, int C, int L, int Ctot, int coff,
                                 long total)
{
    long idx = (long)blockIdx.x * blockDim.x + threadIdx.x;
    if (idx >= total) return;
    int l  = (int)(idx % L);
    long r = idx / L;
    int c  = (int)(r % C);
    long b = r / C;
    out[(b * Ctot + coff + c) * (long)L + l] = in[idx];
}

__global__ void split_heads_kernel(const float* in, float* out, int Bn, int T, int H, int hd,
                                   long total)
{
    long idx = (long)blockIdx.x * blockDim.x + threadIdx.x;
    if (idx >= total) return;
    int p  = (int)(idx % hd);
    long r = idx / hd;
    int t  = (int)(r % T); r /= T;
    int b  = (int)(r % Bn);
    int h  = (int)(r / Bn);
    out[idx] = in[((long)b * T + t) * (long)(H * hd) + h * hd + p];
}

__global__ void merge_heads_kernel(const float* in, float* out, int Bn, int T, int H, int hd,
                                   long total)
{
    long idx = (long)blockIdx.x * blockDim.x + threadIdx.x;
    if (idx >= total) return;
    int p  = (int)(idx % hd);
    long r = idx / hd;
    int h  = (int)(r % H); r /= H;
    int t  = (int)(r % T);
    long b = r / T;
    out[idx] = in[(((long)h * Bn + b) * T + t) * (long)hd + p];
}

// ---------------- diffmamba: d = x1 - lam*x2, LN over (T,hd), gn scale ------
__global__ void diff_groupnorm_kernel(const float* __restrict__ x1, const float* __restrict__ x2,
                                      const float* __restrict__ lam, const float* __restrict__ gnw,
                                      const float* __restrict__ gnb, float* __restrict__ out,
                                      int Bn, int T, int hd)
{
    int hb = blockIdx.x;            // h*Bn + b
    int h  = hb / Bn;
    int tid = threadIdx.x;
    long base = (long)hb * T * hd;
    int total = T * hd;
    float lsum = 0.f, lsq = 0.f;
    for (int i = tid; i < total; i += blockDim.x) {
        int p = i % hd;
        float v = x1[base + i] - lam[h * hd + p] * x2[base + i];
        lsum += v; lsq += v * v;
    }
    __shared__ float s1[256], s2[256];
    s1[tid] = lsum; s2[tid] = lsq;
    __syncthreads();
    for (int s = blockDim.x >> 1; s > 0; s >>= 1) {
        if (tid < s) { s1[tid] += s1[tid + s]; s2[tid] += s2[tid + s]; }
        __syncthreads();
    }
    float mu  = s1[0] / total;
    float var = s2[0] / total - mu * mu;
    float inv = rsqrtf(var + 1e-5f);
    for (int i = tid; i < total; i += blockDim.x) {
        int p = i % hd;
        float v = x1[base + i] - lam[h * hd + p] * x2[base + i];
        out[base + i] = (v - mu) * inv * gnw[h * hd + p] + gnb[h * hd + p];
    }
}

// ---------------- SAIE weights: MLP over pos diffs + softmax over c ---------
__global__ void saie_kernel(const float* in_pos, const float* out_pos,
                            const float* W1, const float* b1,
                            const float* W2, const float* b2, float* w)
{
    int o = blockIdx.x;
    int c = threadIdx.x;
    float d0 = out_pos[o * 3 + 0] - in_pos[c * 3 + 0];
    float d1 = out_pos[o * 3 + 1] - in_pos[c * 3 + 1];
    float d2 = out_pos[o * 3 + 2] - in_pos[c * 3 + 2];
    float acc = b2[0];
    for (int hh = 0; hh < 32; ++hh) {
        float hv = d0 * W1[hh] + d1 * W1[32 + hh] + d2 * W1[64 + hh] + b1[hh];
        acc += fmaxf(hv, 0.f) * W2[hh];
    }
    __shared__ float sh[64];
    sh[c] = acc;
    __syncthreads();
    float m = -1e30f;
    for (int i = 0; i < 64; ++i) m = fmaxf(m, sh[i]);
    __syncthreads();
    float e = __expf(acc - m);
    sh[c] = e;
    __syncthreads();
    float s = 0.f;
    for (int i = 0; i < 64; ++i) s += sh[i];
    w[o * 64 + c] = e / s;
}

// ---------------------------------------------------------------------------
// Host-side GEMM driver: convert A once, pack B once, run branch-free WMMA.
// ---------------------------------------------------------------------------
struct GemmWS { _Float16* AH; _Float16* BP; };

static void gemm(hipStream_t st, GemmWS ws, const float* A, const float* B, const float* bias,
                 float* C, int M, int N, int K, int batch, bool aStrided, bool bStrided, int act)
{
    int tilesN = (N + 15) / 16;
    int kch    = (K + 31) / 32;
    long aTotal = (aStrided ? (long)batch : 1L) * M * K;
    cvt_f16_kernel<<<(unsigned)((aTotal + 255) / 256), 256, 0, st>>>(A, ws.AH, aTotal);
    long sBp = (long)tilesN * kch * 512;
    {
        dim3 g(tilesN, kch, bStrided ? batch : 1);
        pack_b_kernel<<<g, 32, 0, st>>>(B, ws.BP, N, K, kch,
                                        bStrided ? (long)K * N : 0L, sBp);
    }
    {
        dim3 g(tilesN, M / 64, batch);
        wmma_gemm_f16_kernel<<<g, 32, 0, st>>>(ws.AH, ws.BP, bias, C, M, N, K, kch,
                                               aStrided ? (long)M * K : 0L,
                                               bStrided ? sBp : 0L,
                                               (long)M * N, act);
    }
}

// ---------------------------------------------------------------------------
// Host-side Mamba-2 driver (shared by enc1 / dm3 / dm2 / diffmamba teams).
// ---------------------------------------------------------------------------
struct MP { const float *Win, *conv_w, *conv_b, *dt_bias, *A_log, *D, *norm_w, *Wout; };

static void run_mamba2(hipStream_t st, GemmWS gws, const float* u, float* out,
                       int NB, int T, int d_model, int headdim, int H,
                       const MP& p, float* scratch)
{
    const int NS = 32;
    int d_inner  = 2 * d_model;
    int nh       = d_inner / headdim;
    int conv_dim = d_inner + 2 * NS;
    int dinp     = 2 * d_inner + 2 * NS + nh;
    int Bper     = NB / H;
    long rows    = (long)NB * T;
    float* zxb = scratch;
    float* xbc = zxb + rows * dinp;
    float* dts = xbc + rows * conv_dim;
    float* dAa = dts + rows * nh;
    float* yv  = dAa + rows * nh;
    long wS = (H > 1) ? 1L : 0L;

    // zxbcdt = u @ Win (batched over heads)
    gemm(st, gws, u, p.Win, nullptr, zxb, Bper * T, dinp, d_model, H, H > 1, H > 1, 0);
    {   // xBC = silu(depthwise causal conv(xBC))
        long total = rows * conv_dim;
        dwconv_silu_kernel<<<(unsigned)((total + 255) / 256), 256, 0, st>>>(
            zxb + d_inner, (long)dinp, xbc, T, conv_dim, p.conv_w, p.conv_b,
            wS * conv_dim * 4, wS * conv_dim, Bper, total);
    }
    {   // dt / dA
        long total = rows * nh;
        dt_prep_kernel<<<(unsigned)((total + 255) / 256), 256, 0, st>>>(
            zxb, dinp, d_inner + conv_dim, p.dt_bias, p.A_log, dts, dAa,
            T, nh, wS * nh, Bper, total);
    }
    {   // sequential scan
        dim3 g(NB * nh), blk(32, 8);
        ssm_scan_kernel<<<g, blk, 0, st>>>(dAa, dts, xbc, conv_dim, d_inner,
                                           p.D, wS * nh, yv, T, nh, headdim, Bper);
    }
    {   // gate + rmsnorm (in place)
        gate_rmsnorm_kernel<<<(unsigned)rows, d_inner, 0, st>>>(
            yv, zxb, dinp, p.norm_w, wS * d_inner, T, d_inner, Bper);
    }
    // out = y @ Wout (batched over heads)
    gemm(st, gws, yv, p.Wout, nullptr, out, Bper * T, d_model, d_inner, H, H > 1, H > 1, 0);
}

// ---------------------------------------------------------------------------
extern "C" void kernel_launch(void* const* d_in, const int* in_sizes, int n_in,
                              void* d_out, int out_size, void* d_ws, size_t ws_size,
                              hipStream_t stream)
{
    (void)in_sizes; (void)n_in; (void)out_size; (void)ws_size;
    const int Bn = 16, T = 2048;

    int i = 0;
    auto F = [&](void) { return (const float*)d_in[i++]; };
    const float* x       = F();
    const float* in_pos  = F();
    const float* out_pos = F();
    const float *saie_W1 = F(), *saie_b1 = F(), *saie_W2 = F(), *saie_b2 = F();
    const float *mb1_w = F(), *mb1_b = F(), *mb3_w = F(), *mb3_b = F();
    const float *mb7_w = F(), *mb7_b = F(), *mbf_w = F(), *mbf_b = F();
    const float *enc1_W = F(), *enc1_b = F();
    auto readMP = [&](MP& m) {
        m.Win = F(); m.conv_w = F(); m.conv_b = F(); m.dt_bias = F();
        m.A_log = F(); m.D = F(); m.norm_w = F(); m.Wout = F();
    };
    MP enc1_m; readMP(enc1_m);
    const float *enc2_w = F(), *enc2_b = F(), *enc3_w = F(), *enc3_b = F();
    const float *bn1_W = F(), *bn1_b = F();
    MP team1, team2; readMP(team1); readMP(team2);
    const float *lam = F(), *gn_w = F(), *gn_b = F(), *Wo = F(), *bo = F();
    const float *bn2_W = F(), *bn2_b = F();
    const float *dec3_w = F(), *dec3_b = F();
    MP dm3; readMP(dm3);
    const float *dec2_w = F(), *dec2_b = F();
    MP dm2; readMP(dm2);
    const float *dec1_w = F(), *dec1_b = F();
    const float *out_w = F(), *out_b = F();

    // workspace bump layout (floats; every size a multiple of 16 -> 64B align)
    float* ws = (float*)d_ws;
    size_t ofs = 0;
    auto alloc = [&](size_t n) { float* p = ws + ofs; ofs += n; return p; };
    float* W_SAIE = alloc(64 * 64);
    float* PX1 = alloc((size_t)Bn * 64 * 2048);   // encoder skip 1
    float* PX2 = alloc((size_t)Bn * 64 * 1024);   // encoder skip 2
    float* PX3 = alloc((size_t)Bn * 128 * 512);   // encoder skip 3
    float* PBN = alloc((size_t)Bn * 256 * 128);   // bottleneck residual
    float* GA  = alloc(6400000);
    float* GB  = alloc(6400000);
    float* GC  = alloc(4300000);
    float* MS  = alloc(22000000);                 // mamba scratch
    GemmWS gws;
    gws.AH = (_Float16*)alloc(2400000);           // 4.8M halves (max 4.2M + slack)
    gws.BP = (_Float16*)alloc(1200000);           // 2.4M halves (max 2.1M + slack)

    auto ew = [&](long n) { return (unsigned)((n + 255) / 256); };

    // ---- stage 0: SAIE mixing weights -------------------------------------
    saie_kernel<<<64, 64, 0, stream>>>(in_pos, out_pos, saie_W1, saie_b1, saie_W2, saie_b2, W_SAIE);

    // ---- stage 1: x = einsum('oc,bct->bot'): W_SAIE[64x64] @ x[b] ----------
    gemm(stream, gws, W_SAIE, x, nullptr, GA, 64, 2048, 64, Bn, false, true, 0);

    // ---- stage 2: multi-scale convs -> E(192ch), then 1x1 fuse -> x0 ------
    auto conv = [&](const float* in, const float* w, const float* b, float* o,
                    int Ci, int Co, int L, int Kw, int pad, int Ctot, int coff) {
        dim3 g((L + 255) / 256, Co, Bn);
        conv1d_kernel<<<g, 256, 0, stream>>>(in, w, b, o, Ci, L, Kw, pad, Ctot, coff);
    };
    conv(GA, mb1_w, mb1_b, GB, 64, 64, 2048, 1, 0, 192, 0);
    conv(GA, mb3_w, mb3_b, GB, 64, 64, 2048, 3, 1, 192, 64);
    conv(GA, mb7_w, mb7_b, GB, 64, 64, 2048, 7, 3, 192, 128);
    conv(GB, mbf_w, mbf_b, GA, 192, 64, 2048, 1, 0, 64, 0);

    // ---- stage 3: enc1 linear + mamba + residual --------------------------
    long nBT64 = (long)Bn * T * 64;
    bct_to_btc_kernel<<<ew(nBT64), 256, 0, stream>>>(GA, GC, 64, 2048, nBT64);
    gemm(stream, gws, GC, enc1_W, enc1_b, GA, Bn * T, 64, 64, 1, false, false, 0);
    run_mamba2(stream, gws, GA, GC, Bn, T, 64, 64, 1, enc1_m, MS);
    add_kernel<<<ew(nBT64), 256, 0, stream>>>(GA, GC, GC, nBT64);
    btc_to_bct_kernel<<<ew(nBT64), 256, 0, stream>>>(GC, PX1, 64, 2048, nBT64);

    // ---- stage 4: encoder pyramid -----------------------------------------
    maxpool2_kernel<<<ew((long)Bn * 64 * 1024), 256, 0, stream>>>(PX1, GA, 1024, (long)Bn * 64 * 1024);
    conv(GA, enc2_w, enc2_b, PX2, 64, 64, 1024, 3, 1, 64, 0);
    maxpool2_kernel<<<ew((long)Bn * 64 * 512), 256, 0, stream>>>(PX2, GA, 512, (long)Bn * 64 * 512);
    conv(GA, enc3_w, enc3_b, PX3, 64, 128, 512, 3, 1, 128, 0);
    maxpool2_kernel<<<ew((long)Bn * 128 * 256), 256, 0, stream>>>(PX3, GA, 256, (long)Bn * 128 * 256);
    long nBN = (long)Bn * 256 * 128;
    bct_to_btc_kernel<<<ew(nBN), 256, 0, stream>>>(GA, GB, 128, 256, nBN);

    // ---- stage 5: bottleneck bn1 + diffmamba + bn2 ------------------------
    gemm(stream, gws, GB, bn1_W, bn1_b, PBN, Bn * 256, 128, 128, 1, false, false, 0);
    split_heads_kernel<<<ew(nBN), 256, 0, stream>>>(PBN, GA, Bn, 256, 8, 16, nBN);
    float* OUT1 = GB;
    float* OUT2 = GB + 524288;
    run_mamba2(stream, gws, GA, OUT1, 8 * Bn, 256, 16, 16, 8, team1, MS);
    run_mamba2(stream, gws, GA, OUT2, 8 * Bn, 256, 16, 16, 8, team2, MS);
    diff_groupnorm_kernel<<<8 * Bn, 256, 0, stream>>>(OUT1, OUT2, lam, gn_w, gn_b, GC, Bn, 256, 16);
    merge_heads_kernel<<<ew(nBN), 256, 0, stream>>>(GC, GA, Bn, 256, 8, 16, nBN);
    gemm(stream, gws, GA, Wo, bo, GB, Bn * 256, 128, 128, 1, false, false, 0);
    add_kernel<<<ew(nBN), 256, 0, stream>>>(PBN, GB, PBN, nBN);
    gemm(stream, gws, PBN, bn2_W, bn2_b, GA, Bn * 256, 128, 128, 1, false, false, 0);
    btc_to_bct_kernel<<<ew(nBN), 256, 0, stream>>>(GA, GB, 128, 256, nBN);
    interp_kernel<<<ew((long)Bn * 128 * 512), 256, 0, stream>>>(GB, GC, 256, 512, (long)Bn * 128 * 512);

    // ---- stage 6: dec3 + mamba --------------------------------------------
    long n3 = (long)Bn * 128 * 512;
    copy_chan_kernel<<<ew(n3), 256, 0, stream>>>(PX3, GA, 128, 512, 256, 0, n3);
    copy_chan_kernel<<<ew(n3), 256, 0, stream>>>(GC, GA, 128, 512, 256, 128, n3);
    conv(GA, dec3_w, dec3_b, GB, 256, 128, 512, 3, 1, 128, 0);
    bct_to_btc_kernel<<<ew(n3), 256, 0, stream>>>(GB, GC, 128, 512, n3);
    run_mamba2(stream, gws, GC, GA, Bn, 512, 128, 64, 1, dm3, MS);
    add_kernel<<<ew(n3), 256, 0, stream>>>(GC, GA, GC, n3);
    btc_to_bct_kernel<<<ew(n3), 256, 0, stream>>>(GC, GA, 128, 512, n3);
    interp_kernel<<<ew((long)Bn * 128 * 1024), 256, 0, stream>>>(GA, GB, 512, 1024, (long)Bn * 128 * 1024);

    // ---- stage 7: dec2 + mamba --------------------------------------------
    long n2a = (long)Bn * 64 * 1024, n2b = (long)Bn * 128 * 1024;
    copy_chan_kernel<<<ew(n2a), 256, 0, stream>>>(PX2, GA, 64, 1024, 192, 0, n2a);
    copy_chan_kernel<<<ew(n2b), 256, 0, stream>>>(GB, GA, 128, 1024, 192, 64, n2b);
    conv(GA, dec2_w, dec2_b, GC, 192, 64, 1024, 3, 1, 64, 0);
    bct_to_btc_kernel<<<ew(n2a), 256, 0, stream>>>(GC, GA, 64, 1024, n2a);
    run_mamba2(stream, gws, GA, GB, Bn, 1024, 64, 64, 1, dm2, MS);
    add_kernel<<<ew(n2a), 256, 0, stream>>>(GA, GB, GA, n2a);
    btc_to_bct_kernel<<<ew(n2a), 256, 0, stream>>>(GA, GC, 64, 1024, n2a);
    interp_kernel<<<ew(nBT64), 256, 0, stream>>>(GC, GA, 1024, 2048, nBT64);

    // ---- stage 8: dec1 + output head --------------------------------------
    copy_chan_kernel<<<ew(nBT64), 256, 0, stream>>>(PX1, GB, 64, 2048, 128, 0, nBT64);
    copy_chan_kernel<<<ew(nBT64), 256, 0, stream>>>(GA, GB, 64, 2048, 128, 64, nBT64);
    conv(GB, dec1_w, dec1_b, GA, 128, 64, 2048, 3, 1, 64, 0);
    conv(GA, out_w, out_b, (float*)d_out, 64, 64, 2048, 1, 0, 64, 0);
}
